// MultiHeadedAttention_12489764897541
// MI455X (gfx1250) — matrix operand order repaired and checked by reading, not compiled, not run on hardware
//
#include <hip/hip_runtime.h>
#include <hip/hip_bf16.h>

// ---------------------------------------------------------------------------
// MultiHeadedAttention (dual-key masked sigmoid attention) for gfx1250.
// All matmuls run on v_wmma_f32_16x16x32_bf16 (f32 accumulate).
// B=4, S=1024, D=1024, H=16, DK=64.
// Round 6: sigmoid via gfx1250 v_tanh_f32 (sigmoid(x)=0.5+0.5*tanh(x/2)),
// with the /2 folded into the (power-of-two, exact) K0/K1 projection scales.
// Halves the TRANS-pipe work in the attention epilogue.
// ---------------------------------------------------------------------------

#define B_  4
#define S_  1024
#define D_  1024
#define H_  16
#define DK_ 64

typedef __attribute__((ext_vector_type(16))) __bf16 v16bf;
typedef __attribute__((ext_vector_type(8)))  __bf16 v8bf;
typedef __attribute__((ext_vector_type(4)))  __bf16 v4bf;
typedef __attribute__((ext_vector_type(8)))  float  v8f;

__device__ __forceinline__ v16bf join16(v8bf lo, v8bf hi) {
    return __builtin_shufflevector(lo, hi, 0,1,2,3,4,5,6,7,8,9,10,11,12,13,14,15);
}

__device__ __forceinline__ v8f wmma_bf16(v16bf a, v16bf b, v8f c) {
    // (neg_a, A, neg_b, B, c_mod, C, reuse_a, reuse_b)
    return __builtin_amdgcn_wmma_f32_16x16x32_bf16(false, a, false, b, (short)0, c, false, false);
}

__device__ __forceinline__ float tanh_hw(float x) {
#if __has_builtin(__builtin_amdgcn_tanhf)
    return __builtin_amdgcn_tanhf(x);
#elif __has_builtin(__builtin_amdgcn_tanh_f32)
    return __builtin_amdgcn_tanh_f32(x);
#else
    float t;
    // TRANS op: keep one slot before the result is consumed (RAW hazard rule)
    asm("v_tanh_f32 %0, %1\n\tv_nop" : "=v"(t) : "v"(x));
    return t;
#endif
}

// A fragment of a row-major [M x ld] bf16 matrix: 16 rows at m0, K=32 at k0.
// Lane L: row = L&15; elements 0..7 = k0+half*8.., 8..15 = k0+16+half*8..
__device__ __forceinline__ v16bf load_frag_a(const __bf16* __restrict__ X, int ld,
                                             int m0, int k0, int lane) {
    const int row = lane & 15, half = lane >> 4;
    const __bf16* p = X + (size_t)(m0 + row) * ld + k0 + half * 8;
    v8bf lo = *(const v8bf*)(p);
    v8bf hi = *(const v8bf*)(p + 16);
    return join16(lo, hi);
}

// B fragment (32x16) where B[k][n] = W[n][k], W row-major [N x ld].
// Lane L: column n = n0 + (L&15); elements 0..7 = k0+half*8.., 8..15 = k0+16+half*8..
__device__ __forceinline__ v16bf load_frag_b(const __bf16* __restrict__ W, int ld,
                                             int n0, int k0, int lane) {
    const int col = lane & 15, half = lane >> 4;
    const __bf16* p = W + (size_t)(n0 + col) * ld + k0 + half * 8;
    v8bf lo = *(const v8bf*)(p);
    v8bf hi = *(const v8bf*)(p + 16);
    return join16(lo, hi);
}

// ---------------------------------------------------------------------------
// f32 -> bf16 conversion (vec4)
// ---------------------------------------------------------------------------
__global__ void __launch_bounds__(256)
cvt_f32_bf16(const float* __restrict__ src, __bf16* __restrict__ dst, int n4) {
    int i = blockIdx.x * blockDim.x + threadIdx.x;
    if (i < n4) {
        float4 f = ((const float4*)src)[i];
        v4bf o;
        o[0] = (__bf16)f.x; o[1] = (__bf16)f.y; o[2] = (__bf16)f.z; o[3] = (__bf16)f.w;
        ((v4bf*)dst)[i] = o;
    }
}

// ---------------------------------------------------------------------------
// keepT[b][k][q] = (mask0[b][q][k] != 0) & (mask1[b][q][k] != 0)  (byte)
// 32x32 tile transpose through LDS; coalesced loads (along k) and stores
// (along q). grid (S/32, S/32, B), block (32, 8).
// ---------------------------------------------------------------------------
__global__ void __launch_bounds__(256)
combine_mask_T(const int* __restrict__ m0, const int* __restrict__ m1,
               unsigned char* __restrict__ keepT) {
    __shared__ unsigned char tile[32][33];
    const int k0 = blockIdx.x * 32, q0 = blockIdx.y * 32, b = blockIdx.z;
    const int tx = threadIdx.x, ty = threadIdx.y;
    const size_t base = (size_t)b * S_ * S_;
#pragma unroll
    for (int i = 0; i < 4; ++i) {
        const int q = q0 + ty * 4 + i;
        const size_t idx = base + (size_t)q * S_ + k0 + tx;
        tile[ty * 4 + i][tx] = (unsigned char)((m0[idx] != 0) & (m1[idx] != 0));
    }
    __syncthreads();
#pragma unroll
    for (int i = 0; i < 4; ++i) {
        const int k = k0 + ty * 4 + i;
        keepT[base + (size_t)k * S_ + q0 + tx] = tile[tx][ty * 4 + i];
    }
}

// ---------------------------------------------------------------------------
// GEMM: out[m,n] = wscale * sum_k X[m,k] * W[n,k]   (M=4096, N=1024, K=1024)
// MODE 0: bf16 out, head-major [B,H,S,DK]   (Q, K0, K1; K scaled by 2^-n, exact)
// MODE 1: bf16 out, head-major transposed [B,H,DK,S]  (V)
// MODE 2: f32 out, row-major [M,N]          (final output)
// One wave computes a 32x64 tile (two 16-row subtiles share B fragments);
// 4 waves per block cover 32x256.
// ---------------------------------------------------------------------------
template <int MODE>
__global__ void __launch_bounds__(128, 1)
gemm_xwT(const __bf16* __restrict__ X, const __bf16* __restrict__ W,
         __bf16* __restrict__ outb, float* __restrict__ outf, float wscale) {
    const int K = D_;
    const int lane = threadIdx.x & 31;
    const int wave = threadIdx.x >> 5;
    const int m0 = blockIdx.x * 32;
    const int n0 = (blockIdx.y * 4 + wave) * 64;

    v8f acc[2][4];
#pragma unroll
    for (int mt = 0; mt < 2; ++mt)
#pragma unroll
        for (int t = 0; t < 4; ++t) acc[mt][t] = v8f{0};

    for (int k0 = 0; k0 < K; k0 += 32) {
        const v16bf a0 = load_frag_a(X, K, m0,      k0, lane);
        const v16bf a1 = load_frag_a(X, K, m0 + 16, k0, lane);
#pragma unroll
        for (int t = 0; t < 4; ++t) {
            const v16bf b = load_frag_b(W, K, n0 + t * 16, k0, lane);
            acc[0][t] = wmma_bf16(a0, b, acc[0][t]);
            acc[1][t] = wmma_bf16(a1, b, acc[1][t]);
        }
    }

    const int col = lane & 15, hi = lane >> 4;
    const int b_ = m0 >> 10;   // batch (rows stay within one batch: m0 % 1024 <= 992)

#pragma unroll
    for (int mt = 0; mt < 2; ++mt) {
        const int sml = (m0 & (S_ - 1)) + mt * 16;
#pragma unroll
        for (int t = 0; t < 4; ++t) {
            const int n = n0 + t * 16 + col;
            const int h = n >> 6, dk = n & 63;
            if (MODE == 1) {
                // transposed V store: contiguous 8 bf16 per lane
                v8bf pk;
#pragma unroll
                for (int r = 0; r < 8; ++r) pk[r] = (__bf16)acc[mt][t][r];
                __bf16* dst = outb + ((size_t)(b_ * H_ + h) * DK_ + dk) * S_ + sml + 8 * hi;
                *(v8bf*)dst = pk;
            } else {
#pragma unroll
                for (int r = 0; r < 8; ++r) {
                    const int s = sml + r + 8 * hi;
                    const float v = acc[mt][t][r] * ((MODE == 0) ? wscale : 1.0f);
                    if (MODE == 0) {
                        outb[((size_t)(b_ * H_ + h) * S_ + s) * DK_ + dk] = (__bf16)v;
                    } else {
                        outf[(size_t)(b_ * S_ + s) * D_ + n] = v;
                    }
                }
            }
        }
    }
}

// ---------------------------------------------------------------------------
// Fused attention: per (b,h), per 32-query block (one wave, two 16-q subtiles).
// scores = where(m0, qk0*scale, -1e9) + where(m1, qk1*0.5*scale, -1e9)
// In fp32 this is exactly: p = (m0 && m1) ? sigmoid(qk0*scale + qk1*0.5*scale) : 0
// (|qk'| << ulp(1e9); sigmoid(-1e9)=0.)  Half the score scales (1/16, 1/32;
// powers of two -> exact under bf16 rounding) are pre-folded into K0/K1, so
// s0+s1 == x/2 and sigmoid(x) = 0.5 + 0.5*tanh(s0+s1) via gfx1250 v_tanh_f32.
// keep bytes are 0/1 so the mask select is an exact multiply (branchless).
// keepT[b][k][q] gives each lane its 8 keep-flags as one 8-byte load.
// Q/K head-major [B,H,S,DK]; V transposed [B,H,DK,S]; X2 out [B,S,D] bf16.
// ---------------------------------------------------------------------------
__global__ void __launch_bounds__(32, 1)
attn_kernel(const __bf16* __restrict__ Q,
            const __bf16* __restrict__ Ka,
            const __bf16* __restrict__ Kb,
            const __bf16* __restrict__ Vt,
            const unsigned char* __restrict__ keepT,
            __bf16* __restrict__ X2) {
    __shared__ __bf16 Plds[2][16][40];   // two 16q x 32key P tiles, rows padded (banks)

    const int lane = threadIdx.x;
    const int col = lane & 15, hi = lane >> 4;
    const int q0 = blockIdx.x * 32;
    const int bh = blockIdx.y;
    const int b  = bh >> 4, h = bh & 15;

    const __bf16* Qh  = Q  + (size_t)bh * S_ * DK_;
    const __bf16* K0h = Ka + (size_t)bh * S_ * DK_;
    const __bf16* K1h = Kb + (size_t)bh * S_ * DK_;
    const __bf16* Vh  = Vt + (size_t)bh * DK_ * S_;
    const unsigned char* Mb = keepT + (size_t)b * S_ * S_;

    // Q A-fragments for the whole key loop: 2 q-subtiles x 2 K-chunks (DK=64)
    v16bf qa[2][2];
#pragma unroll
    for (int mt = 0; mt < 2; ++mt) {
        qa[mt][0] = load_frag_a(Qh, DK_, q0 + mt * 16, 0,  lane);
        qa[mt][1] = load_frag_a(Qh, DK_, q0 + mt * 16, 32, lane);
    }

    v8f oacc[2][4];
#pragma unroll
    for (int mt = 0; mt < 2; ++mt)
#pragma unroll
        for (int t = 0; t < 4; ++t) oacc[mt][t] = v8f{0};

    for (int k0 = 0; k0 < S_; k0 += 32) {
#pragma unroll
        for (int kt = 0; kt < 2; ++kt) {
            const int nb = k0 + kt * 16;
            // shared K0/K1 B-fragments for both q-subtiles
            const v16bf kb0_0 = load_frag_b(K0h, DK_, nb, 0,  lane);
            const v16bf kb0_1 = load_frag_b(K0h, DK_, nb, 32, lane);
            const v16bf kb1_0 = load_frag_b(K1h, DK_, nb, 0,  lane);
            const v16bf kb1_1 = load_frag_b(K1h, DK_, nb, 32, lane);

            const int key = nb + col;
            const unsigned char* Mrow = Mb + (size_t)key * S_ + q0 + 8 * hi;
            // 8 keep-flag bytes per subtile, one b64 load each (8-byte aligned)
            const uint2 kf0 = *(const uint2*)(Mrow);
            const uint2 kf1 = *(const uint2*)(Mrow + 16);

#pragma unroll
            for (int mt = 0; mt < 2; ++mt) {
                v8f s0 = v8f{0};
                s0 = wmma_bf16(qa[mt][0], kb0_0, s0);
                s0 = wmma_bf16(qa[mt][1], kb0_1, s0);
                v8f s1 = v8f{0};
                s1 = wmma_bf16(qa[mt][0], kb1_0, s1);
                s1 = wmma_bf16(qa[mt][1], kb1_1, s1);

                const uint2 kf = mt ? kf1 : kf0;
#pragma unroll
                for (int r = 0; r < 8; ++r) {
                    const unsigned w = (r < 4) ? kf.x : kf.y;
                    const float keepf = (float)((w >> (8 * (r & 3))) & 0xffu); // 0.0 or 1.0
                    const float t   = tanh_hw(s0[r] + s1[r]);   // x/2 pre-scaled
                    const float p   = keepf * __builtin_fmaf(0.5f, t, 0.5f);
                    Plds[mt][r + 8 * hi][kt * 16 + col] = (__bf16)p;
                }
            }
        }
        // store->load LDS transpose within the wave: drain DS counter explicitly
        asm volatile("s_wait_dscnt 0" ::: "memory");

        // reload P tiles as A-fragments [16 q x 32 keys]
        v16bf pa[2];
#pragma unroll
        for (int mt = 0; mt < 2; ++mt) {
            v8bf plo = *(const v8bf*)(&Plds[mt][col][hi * 8]);
            v8bf phi = *(const v8bf*)(&Plds[mt][col][16 + hi * 8]);
            pa[mt] = join16(plo, phi);
        }

#pragma unroll
        for (int t = 0; t < 4; ++t) {
            // B[k][dk] = Vt[dk][k] -> contiguous along keys; shared by both q-subtiles
            const v16bf vb = load_frag_b(Vh, S_, t * 16, k0, lane);
            oacc[0][t] = wmma_bf16(pa[0], vb, oacc[0][t]);
            oacc[1][t] = wmma_bf16(pa[1], vb, oacc[1][t]);
        }
    }

    // x -> [B, S, D] bf16 for the output projection
#pragma unroll
    for (int mt = 0; mt < 2; ++mt) {
#pragma unroll
        for (int t = 0; t < 4; ++t) {
            const int dk = t * 16 + col;
#pragma unroll
            for (int r = 0; r < 8; ++r) {
                const int q = q0 + mt * 16 + r + 8 * hi;
                X2[((size_t)b * S_ + q) * D_ + h * DK_ + dk] = (__bf16)oacc[mt][t][r];
            }
        }
    }
}

// ---------------------------------------------------------------------------
extern "C" void kernel_launch(void* const* d_in, const int* in_sizes, int n_in,
                              void* d_out, int out_size, void* d_ws, size_t ws_size,
                              hipStream_t stream) {
    (void)in_sizes; (void)n_in; (void)out_size; (void)ws_size;

    const float* query = (const float*)d_in[0];
    const float* key_x = (const float*)d_in[1];
    const float* value = (const float*)d_in[2];
    const int*   mask0 = (const int*)d_in[3];
    const int*   mask1 = (const int*)d_in[4];
    const float* Wq    = (const float*)d_in[5];
    const float* Wk0   = (const float*)d_in[6];
    const float* Wk1   = (const float*)d_in[7];
    const float* Wv    = (const float*)d_in[8];
    const float* Wo    = (const float*)d_in[9];
    float* out = (float*)d_out;

    const size_t ACT  = (size_t)B_ * S_ * D_;   // 4 Mi elements
    const size_t WGT  = (size_t)D_ * D_;        // 1 Mi elements

    char* ws = (char*)d_ws;
    size_t off = 0;
    auto alloc = [&](size_t bytes) { char* p = ws + off; off += bytes; return p; };

    __bf16* qb   = (__bf16*)alloc(ACT * 2);
    __bf16* kb   = (__bf16*)alloc(ACT * 2);
    __bf16* vb   = (__bf16*)alloc(ACT * 2);
    __bf16* Wqb  = (__bf16*)alloc(WGT * 2);
    __bf16* Wk0b = (__bf16*)alloc(WGT * 2);
    __bf16* Wk1b = (__bf16*)alloc(WGT * 2);
    __bf16* Wvb  = (__bf16*)alloc(WGT * 2);
    __bf16* Wob  = (__bf16*)alloc(WGT * 2);
    __bf16* Qh   = (__bf16*)alloc(ACT * 2);   // [B,H,S,DK]
    __bf16* K0h  = (__bf16*)alloc(ACT * 2);
    __bf16* K1h  = (__bf16*)alloc(ACT * 2);
    __bf16* Vth  = (__bf16*)alloc(ACT * 2);   // [B,H,DK,S]
    __bf16* X2   = (__bf16*)alloc(ACT * 2);   // [B,S,D]
    unsigned char* keepT = (unsigned char*)alloc((size_t)B_ * S_ * S_);  // [B,k,q]

    // --- convert f32 -> bf16 ---
    {
        const int TPB = 256;
        int n4a = (int)(ACT / 4), n4w = (int)(WGT / 4);
        cvt_f32_bf16<<<(n4a + TPB - 1) / TPB, TPB, 0, stream>>>(query, qb, n4a);
        cvt_f32_bf16<<<(n4a + TPB - 1) / TPB, TPB, 0, stream>>>(key_x, kb, n4a);
        cvt_f32_bf16<<<(n4a + TPB - 1) / TPB, TPB, 0, stream>>>(value, vb, n4a);
        cvt_f32_bf16<<<(n4w + TPB - 1) / TPB, TPB, 0, stream>>>(Wq,  Wqb,  n4w);
        cvt_f32_bf16<<<(n4w + TPB - 1) / TPB, TPB, 0, stream>>>(Wk0, Wk0b, n4w);
        cvt_f32_bf16<<<(n4w + TPB - 1) / TPB, TPB, 0, stream>>>(Wk1, Wk1b, n4w);
        cvt_f32_bf16<<<(n4w + TPB - 1) / TPB, TPB, 0, stream>>>(Wv,  Wvb,  n4w);
        cvt_f32_bf16<<<(n4w + TPB - 1) / TPB, TPB, 0, stream>>>(Wo,  Wob,  n4w);
    }

    // --- combined transposed byte mask ---
    combine_mask_T<<<dim3(S_ / 32, S_ / 32, B_), dim3(32, 8), 0, stream>>>(mask0, mask1, keepT);

    // --- projections (32-row x 64-col tile per wave, 4 waves/block) ---
    // half score scales (1/16, 1/32) folded into K0/K1: exact (powers of two);
    // attention computes sigmoid(x) = 0.5 + 0.5*tanh(s0+s1)
    dim3 ggrid(B_ * S_ / 32, D_ / (64 * 4));
    dim3 gblk(128);
    gemm_xwT<0><<<ggrid, gblk, 0, stream>>>(qb, Wqb,  Qh,  nullptr, 1.0f);
    gemm_xwT<0><<<ggrid, gblk, 0, stream>>>(kb, Wk0b, K0h, nullptr, 0.0625f);
    gemm_xwT<0><<<ggrid, gblk, 0, stream>>>(kb, Wk1b, K1h, nullptr, 0.03125f);
    gemm_xwT<1><<<ggrid, gblk, 0, stream>>>(vb, Wvb,  Vth, nullptr, 1.0f);

    // --- fused masked-sigmoid attention (32 queries per wave) ---
    dim3 agrid(S_ / 32, B_ * H_);
    attn_kernel<<<agrid, dim3(32), 0, stream>>>(Qh, K0h, K1h, Vth, keepT, X2);

    // --- output projection -> f32 ---
    gemm_xwT<2><<<ggrid, gblk, 0, stream>>>(X2, Wob, nullptr, out, 1.0f);
}
